// ContextInformationCorrectionModule_72172630442497
// MI455X (gfx1250) — matrix operand order repaired and checked by reading, not compiled, not run on hardware
//
#include <hip/hip_runtime.h>
#include <hip/hip_bf16.h>
#include <math.h>

// ---- problem constants (match reference) ----
#define BB 4
#define LL 6400
#define CC 768
#define NHH 6
#define DHH 128
#define NPP 4
#define HS_ 80
#define WS_ 80
#define MM (BB * LL)          // 25600 rows
#define PCOLS 80              // 48 offset cols + 24 aw cols, padded to 80

typedef _Float16 f16;
typedef __attribute__((ext_vector_type(4)))  _Float16 v4h;
typedef __attribute__((ext_vector_type(8)))  _Float16 v8h;
typedef __attribute__((ext_vector_type(16))) _Float16 v16h;
typedef __attribute__((ext_vector_type(8)))  float    v8f;

// ---- gfx1250 async global->LDS copy (ASYNCcnt path), guarded ----
#if __has_builtin(__builtin_amdgcn_global_load_async_to_lds_b128)
#define ASYNC_LDS 1
typedef int v4i_ __attribute__((vector_size(16)));              // matches builtin param
typedef __attribute__((address_space(1))) v4i_ g_v4i;           // global pointer
typedef __attribute__((address_space(3))) v4i_ s_v4i;           // LDS pointer
static __device__ __forceinline__ void async_copy16(const void* g, void* l) {
    __builtin_amdgcn_global_load_async_to_lds_b128(
        (g_v4i*)(unsigned long long)g,                          // full 64-bit VA
        (s_v4i*)(unsigned)(unsigned long long)l,                // low 32b = LDS offset
        0, 0);
}
#else
#define ASYNC_LDS 0
#endif

static __device__ __forceinline__ void wait_async_lds() {
#if ASYNC_LDS
#if __has_builtin(__builtin_amdgcn_s_wait_asynccnt)
    __builtin_amdgcn_s_wait_asynccnt(0);
#else
    asm volatile("s_wait_asynccnt 0x0" ::: "memory");
#endif
#endif
}

// ------------------------------------------------------------------
// Kernel 1: convert / pack weights to f16 once per launch.
// ------------------------------------------------------------------
__global__ __launch_bounds__(256) void pack_weights_kernel(
    const float* __restrict__ Wv, const float* __restrict__ Woff,
    const float* __restrict__ Waw, const float* __restrict__ Wout,
    const float* __restrict__ boff, const float* __restrict__ baw,
    f16* __restrict__ Wv_h, f16* __restrict__ Wout_h,
    f16* __restrict__ Wcat_h, float* __restrict__ bcat)
{
    int i = blockIdx.x * 256 + threadIdx.x;
    if (i < CC * CC) {
        Wv_h[i]   = (f16)Wv[i];
        Wout_h[i] = (f16)Wout[i];
    }
    if (i < CC * PCOLS) {
        int r = i / PCOLS, c = i % PCOLS;
        float v = (c < 48) ? Woff[r * 48 + c]
                : (c < 72) ? Waw[r * 24 + (c - 48)] : 0.0f;
        Wcat_h[i] = (f16)v;
    }
    if (i < PCOLS) {
        bcat[i] = (i < 48) ? boff[i] : (i < 72) ? baw[i - 48] : 0.0f;
    }
}

// ------------------------------------------------------------------
// Kernel 2: fused LayerNorm for x -> q_h (f16) and feat -> f_h (f16).
// ------------------------------------------------------------------
__global__ __launch_bounds__(256) void layernorm_kernel(
    const float* __restrict__ x, const float* __restrict__ feat,
    const float* __restrict__ qw, const float* __restrict__ qb,
    const float* __restrict__ fw, const float* __restrict__ fb,
    f16* __restrict__ q_h, f16* __restrict__ f_h)
{
    __shared__ float red[256];
    const int r   = blockIdx.x;
    const int tid = threadIdx.x;
    const float* src; const float* w; const float* bb; f16* dst;
    if (r < MM) { src = x    + (size_t)r * CC;      w = qw; bb = qb; dst = q_h + (size_t)r * CC; }
    else        { src = feat + (size_t)(r-MM) * CC; w = fw; bb = fb; dst = f_h + (size_t)(r-MM) * CC; }

    float v0 = src[tid], v1 = src[tid + 256], v2 = src[tid + 512];
    red[tid] = v0 + v1 + v2; __syncthreads();
    for (int off = 128; off > 0; off >>= 1) {
        if (tid < off) red[tid] += red[tid + off];
        __syncthreads();
    }
    const float mean = red[0] * (1.0f / CC);
    __syncthreads();

    float d0 = v0 - mean, d1 = v1 - mean, d2 = v2 - mean;
    red[tid] = d0 * d0 + d1 * d1 + d2 * d2; __syncthreads();
    for (int off = 128; off > 0; off >>= 1) {
        if (tid < off) red[tid] += red[tid + off];
        __syncthreads();
    }
    const float rs = rsqrtf(red[0] * (1.0f / CC) + 1e-6f);
    dst[tid]       = (f16)(d0 * rs * w[tid]       + bb[tid]);
    dst[tid + 256] = (f16)(d1 * rs * w[tid + 256] + bb[tid + 256]);
    dst[tid + 512] = (f16)(d2 * rs * w[tid + 512] + bb[tid + 512]);
}

// ------------------------------------------------------------------
// WMMA GEMM: C[M x N] = A[M x K](f16) * B[K x N](f16) + bias, K%64==0.
// Block = 256 threads = 8 waves. Tile BM=128 x BN (64 or 80) x BK=64.
// LDS double-buffered software pipeline:
//   stage t: issue async A(t+1) + load B(t+1) into regs,
//            WMMA-compute tile t, scatter B(t+1) to LDS,
//            s_wait_asynccnt 0, single barrier.
// WMMA chain issued in reverse fragment order so all NT B-fragments
// stay live -> one dscnt wait, then back-to-back v_wmma.
// EPI: 0 -> f16 store, 1 -> f32 store, 2 -> f32 store out = x + gamma*val.
// ------------------------------------------------------------------
template<int BN, int EPI>
__global__ __launch_bounds__(256) void gemm_wmma_kernel(
    const f16* __restrict__ A, const f16* __restrict__ Bm,
    const float* __restrict__ bias, void* __restrict__ Cout,
    int N, int K,
    const float* __restrict__ xres, const float* __restrict__ gamma)
{
    constexpr int BM = 128, BKS = 64, LDP = BKS + 8;   // pad keeps 16B align
    constexpr int NT   = BN / 16;
    constexpr int ACH  = (BM * BKS) / 8;               // A 16B chunks = 1024
    constexpr int ACPT = ACH / 256;                    // 4 per thread
    constexpr int BCH  = (BKS * BN) / 8;               // B 16B chunks
    constexpr int BCPT = (BCH + 255) / 256;            // 2 (BN=64) or 3 (BN=80)
    constexpr int BCPR = BN / 8;                       // B chunks per k-row
    __shared__ f16 Alds[2][BM][LDP];                   // row-major A tiles
    __shared__ f16 Blds[2][BN][LDP];                   // N-major (transposed) B

    const int tid  = threadIdx.x;
    const int lane = tid & 31;
    const int wave = tid >> 5;
    const int m0   = blockIdx.x * BM;
    const int n0   = blockIdx.y * BN;
    const int rlo  = lane & 15;
    const bool hi  = lane >= 16;

    v8f acc[NT];
    #pragma unroll
    for (int nt = 0; nt < NT; ++nt)
        #pragma unroll
        for (int i = 0; i < 8; ++i) acc[nt][i] = 0.0f;

    v8h breg[BCPT];

    auto stageA = [&](int buf, int k0) {
        #pragma unroll
        for (int it = 0; it < ACPT; ++it) {
            const int c   = it * 256 + tid;
            const int row = c >> 3, kc = c & 7;
            const f16* g  = A + (size_t)(m0 + row) * K + k0 + kc * 8;
#if ASYNC_LDS
            async_copy16(g, &Alds[buf][row][kc * 8]);
#else
            *(v8h*)&Alds[buf][row][kc * 8] = *(const v8h*)g;
#endif
        }
    };
    auto loadB = [&](int k0) {
        #pragma unroll
        for (int it = 0; it < BCPT; ++it) {
            const int c = it * 256 + tid;
            if (BCH % 256 == 0 || c < BCH) {
                const int kk = c / BCPR, nc = c % BCPR;
                breg[it] = *(const v8h*)(Bm + (size_t)(k0 + kk) * N + n0 + nc * 8);
            }
        }
    };
    auto storeB = [&](int buf) {
        #pragma unroll
        for (int it = 0; it < BCPT; ++it) {
            const int c = it * 256 + tid;
            if (BCH % 256 == 0 || c < BCH) {
                const int kk = c / BCPR, nc = c % BCPR;
                #pragma unroll
                for (int j = 0; j < 8; ++j) Blds[buf][nc * 8 + j][kk] = breg[it][j];
            }
        }
    };
    auto compute = [&](int buf) {
        #pragma unroll
        for (int ks = 0; ks < BKS; ks += 32) {
            // A fragment (16x32): lanes 0-15 K{0..7,16..23}, 16-31 K{8..15,24..31}
            const int arow = wave * 16 + rlo;
            const int ka   = ks + (hi ? 8 : 0);
            v8h alo = *(const v8h*)&Alds[buf][arow][ka];
            v8h ahi = *(const v8h*)&Alds[buf][arow][ka + 16];
            v16h afrag;
            #pragma unroll
            for (int i = 0; i < 8; ++i) { afrag[i] = alo[i]; afrag[i + 8] = ahi[i]; }

            // load all B fragments first (forward order)...
            v16h bfr[NT];
            const int kb = ks + (hi ? 8 : 0);
            #pragma unroll
            for (int nt = 0; nt < NT; ++nt) {
                const int n = nt * 16 + rlo;
                v8h blo = *(const v8h*)&Blds[buf][n][kb];
                v8h bhi = *(const v8h*)&Blds[buf][n][kb + 16];
                #pragma unroll
                for (int i = 0; i < 8; ++i) { bfr[nt][i] = blo[i]; bfr[nt][i + 8] = bhi[i]; }
            }
            // ...then consume in REVERSE order: forces all fragments live,
            // one dscnt wait, back-to-back WMMA issue.
            #pragma unroll
            for (int nt = NT - 1; nt >= 0; --nt)
                acc[nt] = __builtin_amdgcn_wmma_f32_16x16x32_f16(
                    false, afrag, false, bfr[nt], (short)0, acc[nt], false, false);
        }
    };

    // ---- prologue: fill buffer 0 ----
    stageA(0, 0);
    loadB(0);
    storeB(0);
    wait_async_lds();
    __syncthreads();

    // ---- pipelined main loop: one barrier per stage ----
    int buf = 0;
    for (int k0 = 0; k0 < K; k0 += BKS) {
        const bool more = (k0 + BKS) < K;
        if (more) {
            stageA(buf ^ 1, k0 + BKS);   // async copies overlap compute below
            loadB(k0 + BKS);             // global->regs overlaps compute below
        }
        compute(buf);
        if (more) {
            storeB(buf ^ 1);
            wait_async_lds();
            __syncthreads();
        }
        buf ^= 1;
    }

    // --- epilogue: D element v -> M = v + (hi?8:0), N = lane&15 ---
    const int mbase = m0 + wave * 16 + (hi ? 8 : 0);
    #pragma unroll
    for (int nt = 0; nt < NT; ++nt) {
        const int col = n0 + nt * 16 + rlo;
        const float bv = bias ? bias[col] : 0.0f;
        #pragma unroll
        for (int v = 0; v < 8; ++v) {
            const size_t idx = (size_t)(mbase + v) * N + col;
            const float val = acc[nt][v] + bv;
            if (EPI == 0)      ((f16*)Cout)[idx]   = (f16)val;
            else if (EPI == 1) ((float*)Cout)[idx] = val;
            else               ((float*)Cout)[idx] = xres[idx] + gamma[col] * val;
        }
    }
}

// ------------------------------------------------------------------
// Kernel 5: deformable sampling. One wave per (token m, head h).
// ------------------------------------------------------------------
__global__ __launch_bounds__(256) void sample_kernel(
    const float* __restrict__ ref, const float* __restrict__ P,
    const f16* __restrict__ value_h, f16* __restrict__ samp_h)
{
    const int wid  = blockIdx.x * 8 + (threadIdx.x >> 5);
    const int lane = threadIdx.x & 31;
    const int m = wid / NHH;
    const int h = wid % NHH;
    const int b = m / LL;

    const float rpx = ref[(size_t)m * 2 + 0];
    const float rpy = ref[(size_t)m * 2 + 1];
    const float* Pr = P + (size_t)m * PCOLS;

    float lg[NPP], mx = -1e30f;
    #pragma unroll
    for (int p = 0; p < NPP; ++p) { lg[p] = Pr[48 + h * NPP + p]; mx = fmaxf(mx, lg[p]); }
    float se = 0.0f, wts[NPP];
    #pragma unroll
    for (int p = 0; p < NPP; ++p) { wts[p] = __expf(lg[p] - mx); se += wts[p]; }
    const float inv = 1.0f / se;

    const int dbase = h * DHH + lane * 4;
    float acc[4] = {0.0f, 0.0f, 0.0f, 0.0f};

    #pragma unroll
    for (int p = 0; p < NPP; ++p) {
        const float ox = Pr[h * 8 + p * 2 + 0];
        const float oy = Pr[h * 8 + p * 2 + 1];
        const float lx = (rpx + ox * (1.0f / WS_)) * WS_ - 0.5f;
        const float ly = (rpy + oy * (1.0f / HS_)) * HS_ - 0.5f;
        const float x0f = floorf(lx), y0f = floorf(ly);
        const float tx = lx - x0f, ty = ly - y0f;
        const int x0 = (int)x0f, y0 = (int)y0f;
        const float aw = wts[p] * inv;
        #pragma unroll
        for (int dy = 0; dy < 2; ++dy) {
            #pragma unroll
            for (int dx = 0; dx < 2; ++dx) {
                const int xi = x0 + dx, yi = y0 + dy;
                if (xi < 0 || xi >= WS_ || yi < 0 || yi >= HS_) continue; // uniform
                const float wgt = (dx ? tx : 1.0f - tx) * (dy ? ty : 1.0f - ty) * aw;
                const size_t row = (size_t)b * LL + (size_t)yi * WS_ + xi;
                v4h vv = *(const v4h*)(value_h + row * CC + dbase);
                #pragma unroll
                for (int j = 0; j < 4; ++j) acc[j] += wgt * (float)vv[j];
            }
        }
    }
    v4h out;
    #pragma unroll
    for (int j = 0; j < 4; ++j) out[j] = (f16)acc[j];
    *(v4h*)(samp_h + (size_t)m * CC + dbase) = out;
}

// ------------------------------------------------------------------
// Launch
// ------------------------------------------------------------------
extern "C" void kernel_launch(void* const* d_in, const int* in_sizes, int n_in,
                              void* d_out, int out_size, void* d_ws, size_t ws_size,
                              hipStream_t stream)
{
    const float* x     = (const float*)d_in[0];
    const float* refp  = (const float*)d_in[1];
    const float* feat  = (const float*)d_in[2];
    // d_in[3]=spatial_shapes, d_in[4]=level_start_index (static: 80x80, start 0)
    const float* qn_w  = (const float*)d_in[5];
    const float* qn_b  = (const float*)d_in[6];
    const float* fn_w  = (const float*)d_in[7];
    const float* fn_b  = (const float*)d_in[8];
    const float* gamma = (const float*)d_in[9];
    const float* Wv    = (const float*)d_in[10];
    const float* bv    = (const float*)d_in[11];
    const float* Woff  = (const float*)d_in[12];
    const float* boff  = (const float*)d_in[13];
    const float* Waw   = (const float*)d_in[14];
    const float* baw   = (const float*)d_in[15];
    const float* Wout  = (const float*)d_in[16];
    const float* bout  = (const float*)d_in[17];

    char* ws = (char*)d_ws;
    size_t off = 0;
    auto carve = [&](size_t bytes) { char* p = ws + off; off += (bytes + 255) & ~(size_t)255; return p; };
    f16*   q_h     = (f16*)  carve((size_t)MM * CC * sizeof(f16));
    f16*   f_h     = (f16*)  carve((size_t)MM * CC * sizeof(f16));
    f16*   samp_h  = (f16*)  carve((size_t)MM * CC * sizeof(f16));
    f16*   value_h = (f16*)  carve((size_t)MM * CC * sizeof(f16));
    float* P       = (float*)carve((size_t)MM * PCOLS * sizeof(float));
    f16*   Wv_h    = (f16*)  carve((size_t)CC * CC * sizeof(f16));
    f16*   Wout_h  = (f16*)  carve((size_t)CC * CC * sizeof(f16));
    f16*   Wcat_h  = (f16*)  carve((size_t)CC * PCOLS * sizeof(f16));
    float* bcat    = (float*)carve((size_t)PCOLS * sizeof(float));
    (void)ws_size;

    pack_weights_kernel<<<(CC * CC + 255) / 256, 256, 0, stream>>>(
        Wv, Woff, Waw, Wout, boff, baw, Wv_h, Wout_h, Wcat_h, bcat);

    layernorm_kernel<<<2 * MM, 256, 0, stream>>>(
        x, feat, qn_w, qn_b, fn_w, fn_b, q_h, f_h);

    // value = f @ Wv + bv  (f16 out)
    gemm_wmma_kernel<64, 0><<<dim3(MM / 128, CC / 64), 256, 0, stream>>>(
        f_h, Wv_h, bv, value_h, CC, CC, nullptr, nullptr);

    // P = q @ [Woff|Waw] + [boff|baw]  (f32 out)
    gemm_wmma_kernel<80, 1><<<dim3(MM / 128, 1), 256, 0, stream>>>(
        q_h, Wcat_h, bcat, P, PCOLS, CC, nullptr, nullptr);

    // softmax + bilinear gather + point-weighted sum -> samp (f16)
    sample_kernel<<<(MM * NHH) / 8, 256, 0, stream>>>(refp, P, value_h, samp_h);

    // out = x + gamma * (samp @ Wout + bout)
    gemm_wmma_kernel<64, 2><<<dim3(MM / 128, CC / 64), 256, 0, stream>>>(
        samp_h, Wout_h, bout, (float*)d_out, CC, CC, x, gamma);
}
